// GatedGraphConvCNN_21818433864352
// MI455X (gfx1250) — compile-verified, matrix operand order checked by reading, not run on hardware
//
#include <hip/hip_runtime.h>
#include <hip/hip_bf16.h>

// ---------------------------------------------------------------------------
// GatedGraphConv (2 layers, FHC=256) + GRUCell + Conv1d(k=1) + MLP readout
// MI455X / gfx1250: wave32, WMMA f32_16x16x32_f16 for all GEMMs.
// GEMM: each wave owns a 16x64 C strip -> 1 A fragment feeds 4 WMMAs.
// ---------------------------------------------------------------------------

typedef __attribute__((ext_vector_type(16))) _Float16 v16h;
typedef __attribute__((ext_vector_type(8)))  _Float16 v8h;
typedef __attribute__((ext_vector_type(8)))  float    v8f;

#define N_NODES 65536
#define N_EDGES 524288
#define IN_CH   128
#define FHC     256
#define HID     256
#define NCLS    3
#define NPG     32   // nodes per graph

// ---------------------------------------------------------------------------
// Weight prep: gg_w[2][256][256] f32 -> ggT f16 [2][n][k] (transposed so the
// GEMM's B operand is N-major = contiguous K per lane). w_ih/w_hh [768][256]
// are already stored [N][K] for D = A x W^T, so just convert to f16.
// ---------------------------------------------------------------------------
__global__ void prep_weights_k(const float* __restrict__ gg_w,
                               const float* __restrict__ w_ih,
                               const float* __restrict__ w_hh,
                               _Float16* __restrict__ ggT,
                               _Float16* __restrict__ wih16,
                               _Float16* __restrict__ whh16) {
  const int GG = 2 * FHC * FHC;     // 131072
  const int WI = 3 * FHC * FHC;     // 196608 (768*256)
  int idx = blockIdx.x * blockDim.x + threadIdx.x;
  if (idx < GG) {
    int i   = idx >> 16;            // layer
    int rem = idx & 65535;
    int n   = rem >> 8;
    int k   = rem & 255;
    ggT[idx] = (_Float16)gg_w[(size_t)i * (FHC * FHC) + (size_t)k * FHC + n];
  } else if (idx < GG + WI) {
    int j = idx - GG;
    wih16[j] = (_Float16)w_ih[j];
  } else if (idx < GG + 2 * WI) {
    int j = idx - GG - WI;
    whh16[j] = (_Float16)w_hh[j];
  }
}

// ---------------------------------------------------------------------------
// h = pad(x, [N,128] -> [N,256]); also emit f16 copy for WMMA A operand.
// ---------------------------------------------------------------------------
__global__ void pad_x_k(const float* __restrict__ x,
                        float* __restrict__ h, _Float16* __restrict__ h16) {
  int idx = blockIdx.x * blockDim.x + threadIdx.x;
  if (idx >= N_NODES * FHC) return;
  int node = idx >> 8, c = idx & 255;
  float v = (c < IN_CH) ? x[(size_t)node * IN_CH + c] : 0.0f;
  h[idx]   = v;
  h16[idx] = (_Float16)v;
}

__global__ void zero_f32_k(float* __restrict__ p, int n) {
  int i = blockIdx.x * blockDim.x + threadIdx.x;
  if (i < n) p[i] = 0.0f;
}

__global__ void f32_to_f16_k(const float* __restrict__ a,
                             _Float16* __restrict__ b, int n) {
  int i = blockIdx.x * blockDim.x + threadIdx.x;
  if (i < n) b[i] = (_Float16)a[i];
}

// ---------------------------------------------------------------------------
// Fragment load: 16-bit operand layout per CDNA5 ISA 7.12.2.
// lanes 0-15: row = lane,    K = kOff+0..7 (V0-3) and kOff+16..23 (V4-7)
// lanes 16-31: row = lane-16, kOff = 8 -> K = 8..15 and 24..31.
// Two contiguous 16-byte loads per fragment.
// ---------------------------------------------------------------------------
__device__ __forceinline__ v16h load_frag(const _Float16* __restrict__ p) {
  v8h lo = *(const v8h*)p;
  v8h hi = *(const v8h*)(p + 16);
  v16h r;
#pragma unroll
  for (int i = 0; i < 8; ++i) { r[i] = lo[i]; r[8 + i] = hi[i]; }
  return r;
}

__device__ __forceinline__ v8f wmma_f16(v16h a, v16h b, v8f c) {
  return __builtin_amdgcn_wmma_f32_16x16x32_f16(
      /*neg_a=*/false, a, /*neg_b=*/false, b,
      /*c_mod=*/(short)0, c, /*reuse_a=*/false, /*reuse_b=*/false);
}

// ---------------------------------------------------------------------------
// WMMA GEMM: C[M,N] f32 = A[M,K] f16 (row major) x Bt[N,K] f16 (N-major, i.e.
// C[m,n] = sum_k A[m,k]*Bt[n,k]).  Each wave computes a 16x64 C strip:
// per K-step one A fragment is reused across 4 B fragments (4 WMMAs),
// cutting A traffic 4x vs. a 16x16-per-wave tiling.
// C f32 layout: lane -> N = lane&15; VGPR r -> M = r + 8*(lane>=16).
// N must be a multiple of 64, K a multiple of 32.
// ---------------------------------------------------------------------------
__global__ __launch_bounds__(128) void gemm_wmma_f16_k(
    const _Float16* __restrict__ A, const _Float16* __restrict__ Bt,
    float* __restrict__ C, int M, int N, int K) {
  const int lane = threadIdx.x & 31;
  const int wave = threadIdx.x >> 5;
  const int ntn  = N >> 6;                      // 64-wide strips per row
  const int tile = blockIdx.x * 4 + wave;
  const int mt = (tile / ntn) << 4;
  const int nt = (tile % ntn) << 6;
  if (mt >= M) return;

  const int l15  = lane & 15;
  const int kOff = (lane >> 4) << 3;            // 0 or 8
  const _Float16* arow = A  + (size_t)(mt + l15) * K + kOff;
  const _Float16* brow = Bt + (size_t)(nt + l15) * K + kOff;
  const size_t bstep = (size_t)16 * K;          // next 16-column B panel

  v8f acc0 = {}, acc1 = {}, acc2 = {}, acc3 = {};
  for (int kb = 0; kb < K; kb += 32) {
    v16h a  = load_frag(arow + kb);
    v16h b0 = load_frag(brow + kb);
    v16h b1 = load_frag(brow + bstep + kb);
    v16h b2 = load_frag(brow + 2 * bstep + kb);
    v16h b3 = load_frag(brow + 3 * bstep + kb);
    // Pull the next A K-slab toward GL2 (global_prefetch_b8).
    __builtin_prefetch(arow + kb + 128, 0, 0);
    acc0 = wmma_f16(a, b0, acc0);
    acc1 = wmma_f16(a, b1, acc1);
    acc2 = wmma_f16(a, b2, acc2);
    acc3 = wmma_f16(a, b3, acc3);
  }

  float* crow = C + (size_t)(mt + ((lane >> 4) << 3)) * N + nt + l15;
#pragma unroll
  for (int r = 0; r < 8; ++r) {
    crow[(size_t)r * N +  0] = acc0[r];
    crow[(size_t)r * N + 16] = acc1[r];
    crow[(size_t)r * N + 32] = acc2[r];
    crow[(size_t)r * N + 48] = acc3[r];
  }
}

// ---------------------------------------------------------------------------
// Edge gather * edge_weight -> scatter-add into agg (f32 atomics).
// 4 edges per 256-thread block; 64 lanes/edge, float4 per lane (coalesced).
// ---------------------------------------------------------------------------
__global__ __launch_bounds__(256) void scatter_edges_k(
    const float* __restrict__ m, const int* __restrict__ src,
    const int* __restrict__ dst, const float* __restrict__ ew,
    float* __restrict__ agg) {
  int e = blockIdx.x * 4 + (threadIdx.x >> 6);
  if (e >= N_EDGES) return;
  int cg = (threadIdx.x & 63) << 2;
  int s = src[e], d = dst[e];
  float w = ew[e];
  const float4 v = *(const float4*)(m + (size_t)s * FHC + cg);
  float* a = agg + (size_t)d * FHC + cg;
  atomicAdd(a + 0, v.x * w);
  atomicAdd(a + 1, v.y * w);
  atomicAdd(a + 2, v.z * w);
  atomicAdd(a + 3, v.w * w);
}

// ---------------------------------------------------------------------------
// Fused GRUCell elementwise (biases folded here): torch gate order r,z,n.
// Writes new h (f32) and f16 copy for the next layer's WMMA A operand.
// ---------------------------------------------------------------------------
__global__ void gru_update_k(const float* __restrict__ gi,
                             const float* __restrict__ gh,
                             const float* __restrict__ b_ih,
                             const float* __restrict__ b_hh,
                             float* __restrict__ h,
                             _Float16* __restrict__ h16) {
  int idx = blockIdx.x * blockDim.x + threadIdx.x;
  if (idx >= N_NODES * FHC) return;
  int node = idx >> 8, c = idx & 255;
  size_t gb = (size_t)node * (3 * FHC);
  float ir = gi[gb + c]           + b_ih[c];
  float iz = gi[gb + FHC + c]     + b_ih[FHC + c];
  float ig = gi[gb + 2 * FHC + c] + b_ih[2 * FHC + c];
  float hr = gh[gb + c]           + b_hh[c];
  float hz = gh[gb + FHC + c]     + b_hh[FHC + c];
  float hg = gh[gb + 2 * FHC + c] + b_hh[2 * FHC + c];
  float hv = h[idx];
  float r = 1.0f / (1.0f + __expf(-(ir + hr)));
  float z = 1.0f / (1.0f + __expf(-(iz + hz)));
  float g = tanhf(ig + r * hg);
  float hn = (1.0f - z) * g + z * hv;
  h[idx]   = hn;
  h16[idx] = (_Float16)hn;
}

// ---------------------------------------------------------------------------
// Readout: one block per graph. relu(h) -> Conv1d(256->1,k=1) -> relu ->
// Linear(32->256)+relu -> Linear(256->3) -> softmax.
// ---------------------------------------------------------------------------
__global__ __launch_bounds__(256) void readout_k(
    const float* __restrict__ h, const float* __restrict__ cnn_w,
    const float* __restrict__ cnn_b, const float* __restrict__ l1w,
    const float* __restrict__ l1b, const float* __restrict__ l2w,
    const float* __restrict__ l2b, float* __restrict__ out) {
  __shared__ float part[256];
  __shared__ float yv[NPG];
  __shared__ float zs[HID];
  __shared__ float r0[256], r1[256], r2[256];

  const int b = blockIdx.x;
  const int t = threadIdx.x;
  const int e = t >> 3, p = t & 7;     // node-in-graph, channel slice

  // y[b,e] = relu( sum_f relu(h[b,e,f]) * cnn_w[f] + cnn_b )
  const float* hrow = h + ((size_t)b * NPG + e) * FHC + p * 32;
  float s = 0.0f;
#pragma unroll 8
  for (int c = 0; c < 32; ++c) {
    float v = hrow[c];
    v = v > 0.0f ? v : 0.0f;
    s += v * cnn_w[p * 32 + c];
  }
  part[t] = s;
  __syncthreads();
  if (p == 0) {
    float acc = cnn_b[0];
#pragma unroll
    for (int q = 0; q < 8; ++q) acc += part[e * 8 + q];
    yv[e] = acc > 0.0f ? acc : 0.0f;
  }
  __syncthreads();

  // z[t] = relu( lin1_b[t] + sum_e lin1_w[t,e] * y[e] )
  float zacc = l1b[t];
#pragma unroll 8
  for (int ee = 0; ee < NPG; ++ee) zacc += l1w[t * NPG + ee] * yv[ee];
  zs[t] = zacc > 0.0f ? zacc : 0.0f;
  __syncthreads();

  // o[c] = lin2_b[c] + sum_t lin2_w[c,t] * z[t]
  float zz = zs[t];
  r0[t] = l2w[t] * zz;
  r1[t] = l2w[HID + t] * zz;
  r2[t] = l2w[2 * HID + t] * zz;
  __syncthreads();
  for (int st = 128; st > 0; st >>= 1) {
    if (t < st) { r0[t] += r0[t + st]; r1[t] += r1[t + st]; r2[t] += r2[t + st]; }
    __syncthreads();
  }
  if (t == 0) {
    float o0 = r0[0] + l2b[0], o1 = r1[0] + l2b[1], o2 = r2[0] + l2b[2];
    float mx = fmaxf(o0, fmaxf(o1, o2));
    float e0 = __expf(o0 - mx), e1 = __expf(o1 - mx), e2 = __expf(o2 - mx);
    float inv = 1.0f / (e0 + e1 + e2);
    out[b * 3 + 0] = e0 * inv;
    out[b * 3 + 1] = e1 * inv;
    out[b * 3 + 2] = e2 * inv;
  }
}

// ---------------------------------------------------------------------------
// Host-side orchestration.
// ---------------------------------------------------------------------------
extern "C" void kernel_launch(void* const* d_in, const int* in_sizes, int n_in,
                              void* d_out, int out_size, void* d_ws, size_t ws_size,
                              hipStream_t stream) {
  (void)in_sizes; (void)n_in; (void)out_size; (void)ws_size;

  const float* x        = (const float*)d_in[0];
  const int*   ei       = (const int*)d_in[1];     // [2, E]: src row then dst row
  const float* eattr    = (const float*)d_in[2];
  const float* gg_w     = (const float*)d_in[3];
  const float* w_ih     = (const float*)d_in[4];
  const float* w_hh     = (const float*)d_in[5];
  const float* b_ih     = (const float*)d_in[6];
  const float* b_hh     = (const float*)d_in[7];
  const float* cnn_w    = (const float*)d_in[8];
  const float* cnn_b    = (const float*)d_in[9];
  const float* lin1_w   = (const float*)d_in[10];
  const float* lin1_b   = (const float*)d_in[11];
  const float* lin2_w   = (const float*)d_in[12];
  const float* lin2_b   = (const float*)d_in[13];
  float* out = (float*)d_out;

  // Workspace carve-up (bytes).
  char* ws = (char*)d_ws;
  const size_t SZ_H32   = (size_t)N_NODES * FHC * 4;        // 64 MB
  const size_t SZ_H16   = (size_t)N_NODES * FHC * 2;        // 32 MB
  const size_t SZ_G32   = (size_t)N_NODES * 3 * FHC * 4;    // 192 MB
  float*    h     = (float*)(ws);                    ws += SZ_H32;
  _Float16* h16   = (_Float16*)(ws);                 ws += SZ_H16;
  float*    m     = (float*)(ws);                    ws += SZ_H32;
  float*    agg   = (float*)(ws);                    ws += SZ_H32;
  _Float16* agg16 = (_Float16*)(ws);                 ws += SZ_H16;
  float*    gi    = (float*)(ws);                    ws += SZ_G32;
  float*    gh    = (float*)(ws);                    ws += SZ_G32;
  _Float16* ggT   = (_Float16*)(ws);                 ws += (size_t)2 * FHC * FHC * 2;
  _Float16* wih16 = (_Float16*)(ws);                 ws += (size_t)3 * FHC * FHC * 2;
  _Float16* whh16 = (_Float16*)(ws);                 ws += (size_t)3 * FHC * FHC * 2;

  const int NE = N_NODES * FHC;                      // 16,777,216 elements

  // 1) Weight conversion / transpose (f32 -> f16).
  {
    int total = 2 * FHC * FHC + 2 * (3 * FHC * FHC);  // 524288
    prep_weights_k<<<(total + 255) / 256, 256, 0, stream>>>(
        gg_w, w_ih, w_hh, ggT, wih16, whh16);
  }
  // 2) Pad x into h (f32 + f16).
  pad_x_k<<<(NE + 255) / 256, 256, 0, stream>>>(x, h, h16);

  const int* src = ei;
  const int* dst = ei + N_EDGES;

  for (int layer = 0; layer < 2; ++layer) {
    // m = h @ gg_w[layer]   (M=65536, N=256, K=256); 16x64 strips, 4 waves/blk
    {
      int strips = (N_NODES / 16) * (FHC / 64);       // 16384
      gemm_wmma_f16_k<<<strips / 4, 128, 0, stream>>>(
          h16, ggT + (size_t)layer * FHC * FHC, m, N_NODES, FHC, FHC);
    }
    // agg = scatter_add(m[src] * w_e, dst)
    zero_f32_k<<<(NE + 255) / 256, 256, 0, stream>>>(agg, NE);
    scatter_edges_k<<<N_EDGES / 4, 256, 0, stream>>>(m, src, dst, eattr, agg);
    f32_to_f16_k<<<(NE + 255) / 256, 256, 0, stream>>>(agg, agg16, NE);
    // gi = agg @ w_ih^T, gh = h @ w_hh^T   (M=65536, N=768, K=256)
    {
      int strips = (N_NODES / 16) * ((3 * FHC) / 64); // 49152
      gemm_wmma_f16_k<<<strips / 4, 128, 0, stream>>>(
          agg16, wih16, gi, N_NODES, 3 * FHC, FHC);
      gemm_wmma_f16_k<<<strips / 4, 128, 0, stream>>>(
          h16, whh16, gh, N_NODES, 3 * FHC, FHC);
    }
    // GRU elementwise update.
    gru_update_k<<<(NE + 255) / 256, 256, 0, stream>>>(gi, gh, b_ih, b_hh, h, h16);
  }

  // Readout: 2048 graphs, one block each.
  readout_k<<<N_NODES / NPG, 256, 0, stream>>>(
      h, cnn_w, cnn_b, lin1_w, lin1_b, lin2_w, lin2_b, out);
}
